// UNetAttentionShard_19928648253772
// MI455X (gfx1250) — matrix operand order repaired and checked by reading, not compiled, not run on hardware
//
#include <hip/hip_runtime.h>
#include <hip/hip_bf16.h>
#include <stdint.h>

typedef __attribute__((ext_vector_type(16))) __bf16 v16bf;
typedef __attribute__((ext_vector_type(8)))  float  v8f;
typedef __attribute__((ext_vector_type(4)))  unsigned int u32x4;
typedef __attribute__((ext_vector_type(8)))  int i32x8;
typedef __attribute__((ext_vector_type(4)))  int i32x4;

#define WMMA_BF16(a, b, c) \
  __builtin_amdgcn_wmma_f32_16x16x32_bf16(false, (a), false, (b), (short)0, (c), false, false)

// ---- Tensor Data Mover: 2D bf16 tile (strided rows -> packed LDS) ----------
static __device__ __forceinline__ void tdm_load_2d_bf16(
    unsigned int ldsByteAddr, const void* gptr,
    unsigned int widthElems, unsigned int heightRows,
    unsigned long long rowStrideElems) {
  const unsigned long long ga = (unsigned long long)(uintptr_t)gptr;
  u32x4 g0;
  g0[0] = 0x1u;                                   // count=1, user descriptor
  g0[1] = ldsByteAddr;                            // lds_addr
  g0[2] = (unsigned int)(ga & 0xFFFFFFFFu);       // global_addr lo
  g0[3] = (unsigned int)((ga >> 32) & 0x01FFFFFFu) | (0x2u << 30);  // hi | type=2
  i32x8 g1;
  g1[0] = (int)(0x1u << 16);                      // data_size=1 (2 bytes)
  g1[1] = (int)((widthElems & 0xFFFFu) << 16);    // tensor_dim0 lo16
  g1[2] = (int)((widthElems >> 16) | ((heightRows & 0xFFFFu) << 16));
  g1[3] = (int)((heightRows >> 16) | ((widthElems & 0xFFFFu) << 16));  // | tile_dim0
  g1[4] = (int)(heightRows & 0xFFFFu);            // tile_dim1
  g1[5] = (int)(rowStrideElems & 0xFFFFFFFFull);  // dim0_stride lo32
  g1[6] = (int)((rowStrideElems >> 32) & 0xFFFFull);
  g1[7] = 0;
  i32x4 gz = {0, 0, 0, 0};
#if __clang_major__ >= 23
  i32x8 gz8 = {0, 0, 0, 0, 0, 0, 0, 0};
  __builtin_amdgcn_tensor_load_to_lds(g0, g1, gz, gz, gz8, 0);
#else
  __builtin_amdgcn_tensor_load_to_lds(g0, g1, gz, gz, 0);
#endif
}

static __device__ __forceinline__ unsigned int lds_addr_of(const void* p) {
  return (unsigned int)(uintptr_t)p;
}

// ---- VALU-only 16-lane butterfly reductions via DPP ------------------------
template <int CTRL>
static __device__ __forceinline__ float dpp_f32(float v) {
  return __int_as_float(
      __builtin_amdgcn_mov_dpp(__float_as_int(v), CTRL, 0xF, 0xF, true));
}
static __device__ __forceinline__ float red16_max(float v) {
  v = fmaxf(v, dpp_f32<0xB1>(v));    // quad_perm(1,0,3,2)  : xor 1
  v = fmaxf(v, dpp_f32<0x4E>(v));    // quad_perm(2,3,0,1)  : xor 2
  v = fmaxf(v, dpp_f32<0x141>(v));   // row_half_mirror     : xor 7
  v = fmaxf(v, dpp_f32<0x140>(v));   // row_mirror          : xor 15
  return v;
}
static __device__ __forceinline__ float red16_sum(float v) {
  v += dpp_f32<0xB1>(v);
  v += dpp_f32<0x4E>(v);
  v += dpp_f32<0x141>(v);
  v += dpp_f32<0x140>(v);
  return v;
}

// ---- fragment loaders (ISA 7.12.2 layouts, wave32) -------------------------
static __device__ __forceinline__ v16bf frag_ld_a(const __bf16* base, int ld, int lane) {
  const int row = lane & 15;
  const int k0  = (lane < 16) ? 0 : 8;
  union { uint4 q[2]; v16bf v; } u;
  u.q[0] = *(const uint4*)(base + row * ld + k0);
  u.q[1] = *(const uint4*)(base + row * ld + k0 + 16);
  return u.v;
}
static __device__ __forceinline__ v16bf frag_ld_bT(const __bf16* baseT, int ld, int lane) {
  const int col = lane & 15;
  const int k0  = (lane < 16) ? 0 : 8;
  union { uint4 q[2]; v16bf v; } u;
  u.q[0] = *(const uint4*)(baseT + col * ld + k0);
  u.q[1] = *(const uint4*)(baseT + col * ld + k0 + 16);
  return u.v;
}

// ---- tiled GEMM: C[M,N] = A[M,K] x W[K,N] (+bias) --------------------------
template <bool A_BF16, bool OUT_F32>
__global__ __launch_bounds__(256) void gemm128(const void* __restrict__ Ain,
                                               const float* __restrict__ W,
                                               const float* __restrict__ bias,
                                               void* __restrict__ Cout,
                                               int M, int N, int K, float outScale) {
  __shared__ __attribute__((aligned(16))) __bf16 As[128 * 64];   // [m][k]
  __shared__ __attribute__((aligned(16))) __bf16 BsT[128 * 64];  // [n][k]
  const int tid  = threadIdx.x;
  const int lane = tid & 31;
  const int wave = tid >> 5;
  const int wm   = wave & 1;
  const int wn   = wave >> 1;
  const int mBase = blockIdx.y * 128;
  const int nBase = blockIdx.x * 128;
  const unsigned ldsAs = lds_addr_of(As);

  const v8f vzero = {0, 0, 0, 0, 0, 0, 0, 0};
  v8f acc[4][2];
#pragma unroll
  for (int fm = 0; fm < 4; fm++)
#pragma unroll
    for (int fn = 0; fn < 2; fn++) acc[fm][fn] = vzero;

  for (int kt = 0; kt < K; kt += 64) {
    if (A_BF16) {
      if (wave == 0)
        tdm_load_2d_bf16(ldsAs, (const __bf16*)Ain + (size_t)mBase * K + kt,
                         64u, 128u, (unsigned long long)K);
    } else {
#pragma unroll
      for (int r = 0; r < 8; r++) {
        const int idx = r * 1024 + tid * 4;
        const int row = idx >> 6, col = idx & 63;
        const float* Af = (const float*)Ain;
        float4 f = *(const float4*)(Af + (size_t)(mBase + row) * K + kt + col);
        union { uint2 u; __bf16 b[4]; } cv;
        cv.b[0] = (__bf16)f.x; cv.b[1] = (__bf16)f.y;
        cv.b[2] = (__bf16)f.z; cv.b[3] = (__bf16)f.w;
        *(uint2*)(As + row * 64 + col) = cv.u;
      }
    }
#pragma unroll
    for (int r = 0; r < 8; r++) {
      const int idx = r * 1024 + tid * 4;
      const int kr = idx >> 7, col = idx & 127;
      float4 f = *(const float4*)(W + (size_t)(kt + kr) * N + nBase + col);
      BsT[(col + 0) * 64 + kr] = (__bf16)f.x;
      BsT[(col + 1) * 64 + kr] = (__bf16)f.y;
      BsT[(col + 2) * 64 + kr] = (__bf16)f.z;
      BsT[(col + 3) * 64 + kr] = (__bf16)f.w;
    }
    if (A_BF16 && wave == 0) __builtin_amdgcn_s_wait_tensorcnt(0);
    __syncthreads();

#pragma unroll
    for (int kk = 0; kk < 64; kk += 32) {
      v16bf a[4], bb[2];
#pragma unroll
      for (int fm = 0; fm < 4; fm++)
        a[fm] = frag_ld_a(As + (wm * 64 + fm * 16) * 64 + kk, 64, lane);
#pragma unroll
      for (int fn = 0; fn < 2; fn++)
        bb[fn] = frag_ld_bT(BsT + (wn * 32 + fn * 16) * 64 + kk, 64, lane);
#pragma unroll
      for (int fm = 0; fm < 4; fm++)
#pragma unroll
        for (int fn = 0; fn < 2; fn++)
          acc[fm][fn] = WMMA_BF16(a[fm], bb[fn], acc[fm][fn]);
    }
    __syncthreads();
  }

  const int half8 = (lane >> 4) << 3;
  const int c = lane & 15;
#pragma unroll
  for (int fn = 0; fn < 2; fn++) {
    const int col = nBase + wn * 32 + fn * 16 + c;
    const float bv = OUT_F32 ? bias[col] : 0.0f;
#pragma unroll
    for (int fm = 0; fm < 4; fm++) {
#pragma unroll
      for (int vr = 0; vr < 8; vr++) {
        const int row = mBase + wm * 64 + fm * 16 + vr + half8;
        const float v = acc[fm][fn][vr];
        if (OUT_F32)
          ((float*)Cout)[(size_t)row * N + col] = v + bv;
        else
          ((__bf16*)Cout)[(size_t)row * N + col] = (__bf16)(v * outScale);
      }
    }
  }
}

// ---- flash attention, DH=64, 64-key blocks, base-2 online softmax ----------
// Q is pre-scaled by log2(e)/sqrt(DH) so exponentials are native exp2.
__global__ __launch_bounds__(256) void flash64(const __bf16* __restrict__ Q,
                                               const __bf16* __restrict__ Kin,
                                               const __bf16* __restrict__ Vin,
                                               __bf16* __restrict__ O) {
  constexpr int T = 2048, D = 1024, DH = 64;
  __shared__ __attribute__((aligned(16))) __bf16 Qs[128 * 64];    // [qrow][d]
  __shared__ __attribute__((aligned(16))) __bf16 Ks[64 * 64];     // [key][d]
  __shared__ __attribute__((aligned(16))) __bf16 Vt[64 * 64];     // [d][key]
  __shared__ __attribute__((aligned(16))) __bf16 Ps[8 * 16 * 64]; // per-wave P

  const int tid = threadIdx.x, lane = tid & 31, wave = tid >> 5;
  const int qTile = blockIdx.x, h = blockIdx.y, b = blockIdx.z;
  const size_t rowBase = (size_t)b * T;
  const int colBase = h * DH;
  const int half8 = (lane >> 4) << 3;
  const int c = lane & 15;

  if (wave == 0) {
    tdm_load_2d_bf16(lds_addr_of(Qs),
                     Q + (rowBase + (size_t)qTile * 128) * D + colBase,
                     64u, 128u, (unsigned long long)D);
    __builtin_amdgcn_s_wait_tensorcnt(0);
  }
  __syncthreads();

  v16bf qa[2];
  qa[0] = frag_ld_a(Qs + wave * 16 * 64, 64, lane);
  qa[1] = frag_ld_a(Qs + wave * 16 * 64 + 32, 64, lane);

  const v8f vzero = {0, 0, 0, 0, 0, 0, 0, 0};
  v8f Oacc[4];
#pragma unroll
  for (int fn = 0; fn < 4; fn++) Oacc[fn] = vzero;
  float mrow[8], lrow[8];
#pragma unroll
  for (int vr = 0; vr < 8; vr++) { mrow[vr] = -1e30f; lrow[vr] = 0.0f; }

  const unsigned ldsKs = lds_addr_of(Ks);
  for (int jb = 0; jb < T / 64; jb++) {
    if (wave == 0)  // TDM: 64x64 K tile
      tdm_load_2d_bf16(ldsKs, Kin + (rowBase + (size_t)jb * 64) * D + colBase,
                       64u, 64u, (unsigned long long)D);
    // stage V transposed (manual: TDM cannot transpose)
#pragma unroll
    for (int r = 0; r < 2; r++) {
      const int idx = r * 2048 + tid * 8;
      const int key = idx >> 6, d = idx & 63;
      const size_t g = (rowBase + jb * 64 + key) * D + colBase + d;
      union { uint4 q; __bf16 e[8]; } vv;
      vv.q = *(const uint4*)(Vin + g);
#pragma unroll
      for (int j = 0; j < 8; j++) Vt[(d + j) * 64 + key] = vv.e[j];
      if (jb + 1 < T / 64)
        __builtin_prefetch(Vin + g + (size_t)64 * D, 0, 0);
    }
    if (wave == 0) __builtin_amdgcn_s_wait_tensorcnt(0);
    __syncthreads();

    // S = Q K^T : four 16x16 logits tiles (64 keys), 8 WMMAs
    v8f S[4];
#pragma unroll
    for (int n = 0; n < 4; n++) {
      v16bf kb0 = frag_ld_bT(Ks + n * 16 * 64, 64, lane);
      v16bf kb1 = frag_ld_bT(Ks + n * 16 * 64 + 32, 64, lane);
      v8f s = vzero;
      s = WMMA_BF16(qa[0], kb0, s);
      s = WMMA_BF16(qa[1], kb1, s);
      S[n] = s;
    }

    // online softmax in log2 domain; DPP butterfly reductions (no DS traffic)
    __bf16* pw = Ps + wave * 16 * 64;
    float corr[8];
#pragma unroll
    for (int vr = 0; vr < 8; vr++) {
      float mx = fmaxf(fmaxf(S[0][vr], S[1][vr]), fmaxf(S[2][vr], S[3][vr]));
      mx = red16_max(mx);
      const float mnew = fmaxf(mrow[vr], mx);
      const float cf = exp2f(mrow[vr] - mnew);
      float p[4], rs = 0.0f;
#pragma unroll
      for (int n = 0; n < 4; n++) { p[n] = exp2f(S[n][vr] - mnew); rs += p[n]; }
      rs = red16_sum(rs);
      mrow[vr] = mnew;
      lrow[vr] = lrow[vr] * cf + rs;
      corr[vr] = cf;
      const int prow = vr + half8;
#pragma unroll
      for (int n = 0; n < 4; n++) pw[prow * 64 + n * 16 + c] = (__bf16)p[n];
    }
#pragma unroll
    for (int fn = 0; fn < 4; fn++)
#pragma unroll
      for (int vr = 0; vr < 8; vr++) Oacc[fn][vr] *= corr[vr];

    // P x V : 8 WMMAs
    v16bf pa0 = frag_ld_a(pw, 64, lane);
    v16bf pa1 = frag_ld_a(pw + 32, 64, lane);
#pragma unroll
    for (int fn = 0; fn < 4; fn++) {
      v16bf vb0 = frag_ld_bT(Vt + fn * 16 * 64, 64, lane);
      v16bf vb1 = frag_ld_bT(Vt + fn * 16 * 64 + 32, 64, lane);
      Oacc[fn] = WMMA_BF16(pa0, vb0, Oacc[fn]);
      Oacc[fn] = WMMA_BF16(pa1, vb1, Oacc[fn]);
    }
    __syncthreads();
  }

  // normalize and store bf16 attn output in [b*t][h*64+d] layout
#pragma unroll
  for (int vr = 0; vr < 8; vr++) {
    const float inv = 1.0f / lrow[vr];
    const size_t g = rowBase + qTile * 128 + wave * 16 + vr + half8;
#pragma unroll
    for (int fn = 0; fn < 4; fn++)
      O[g * D + colBase + fn * 16 + c] = (__bf16)(Oacc[fn][vr] * inv);
  }
}

// ---- host-side orchestration ----------------------------------------------
extern "C" void kernel_launch(void* const* d_in, const int* in_sizes, int n_in,
                              void* d_out, int out_size, void* d_ws, size_t ws_size,
                              hipStream_t stream) {
  (void)in_sizes; (void)n_in; (void)out_size; (void)ws_size;
  const float* x  = (const float*)d_in[0];
  const float* Wq = (const float*)d_in[1];
  const float* Wk = (const float*)d_in[2];
  const float* Wv = (const float*)d_in[3];
  const float* Wo = (const float*)d_in[4];
  const float* bo = (const float*)d_in[5];

  const int Bb = 4, T = 2048, D = 1024, M = Bb * T;     // 8192
  const size_t plane = (size_t)M * D * sizeof(__bf16);  // 16 MB
  __bf16* Qw = (__bf16*)d_ws;
  __bf16* Kw = (__bf16*)((char*)d_ws + plane);
  __bf16* Vw = (__bf16*)((char*)d_ws + 2 * plane);
  __bf16* Aw = (__bf16*)((char*)d_ws + 3 * plane);

  dim3 blk(256);
  dim3 gProj(D / 128, M / 128);  // (8, 64)
  // Q pre-scaled by log2(e)/sqrt(DH): softmax uses native base-2 exp
  const float qScale = 0.125f * 1.4426950408889634f;
  gemm128<false, false><<<gProj, blk, 0, stream>>>(x, Wq, nullptr, Qw, M, D, D, qScale);
  gemm128<false, false><<<gProj, blk, 0, stream>>>(x, Wk, nullptr, Kw, M, D, D, 1.0f);
  gemm128<false, false><<<gProj, blk, 0, stream>>>(x, Wv, nullptr, Vw, M, D, D, 1.0f);

  dim3 gAttn(T / 128, 16, Bb);  // (16,16,4)
  flash64<<<gAttn, blk, 0, stream>>>(Qw, Kw, Vw, Aw);

  gemm128<true, true><<<gProj, blk, 0, stream>>>(Aw, Wo, bo, (float*)d_out, M, D, D, 1.0f);
}
